// GraphCore_2018634629401
// MI455X (gfx1250) — matrix operand order repaired and checked by reading, not compiled, not run on hardware
//
#include <hip/hip_runtime.h>

// ---------------------------------------------------------------------------
// GAT-like graph layer for MI455X (gfx1250, wave32, WMMA).
// Dominant work: A@A and (A@A)@A with N=1024, B=16  => ~69 GFLOP, done with
// V_WMMA_F32_16X16X32_F16 (exact: adjacency-power values are small integers).
// Both GEMM operands are kept in "fragment-friendly" layout: A row-major,
// B as an explicitly materialized transpose, so every fragment is two
// contiguous global_load_b128 (no strided u16 gathers, no packing moves).
// Each wave computes a 32x32 output block (2x2 WMMA tiles): 4 WMMA per
// 8 x b128 loads per K-step. Working set ~134 MB -> L2-resident (192 MB).
// ---------------------------------------------------------------------------

typedef __attribute__((ext_vector_type(16))) _Float16 v16h;
typedef __attribute__((ext_vector_type(8)))  _Float16 v8h;
typedef __attribute__((ext_vector_type(8)))  float    v8f;

#define GB   16
#define GN   1024
#define GD   8
#define FIN  128
#define FOUT 64
#define LALPHA 0.2f

// A fragment (16-bit 16x32, wave32): lane L: row = L%16, half = L/16
//   elems 0..7  = K[k0 + 8*half + 0..7]        (16B contiguous)
//   elems 8..15 = K[k0 + 16 + 8*half + 0..7]   (16B contiguous)
__device__ __forceinline__ v16h load_frag_A(const _Float16* __restrict__ p, int half) {
  v8h lo = *(const v8h*)(p + 8 * half);
  v8h hi = *(const v8h*)(p + 16 + 8 * half);
  return __builtin_shufflevector(lo, hi, 0, 1, 2, 3, 4, 5, 6, 7,
                                 8, 9, 10, 11, 12, 13, 14, 15);
}

// B fragment from B^T (row-major [n][k]): lane L: col n = L%16,
//   elems 0..7  = BT[n][k0 + 8*half + 0..7], elems 8..15 = BT[n][k0 + 16 + 8*half + 0..7]
//   -> two 16B contiguous chunks in BT row n (same k order as A).
__device__ __forceinline__ v16h load_frag_BT(const _Float16* __restrict__ p, int half) {
  v8h lo = *(const v8h*)(p + 8 * half);
  v8h hi = *(const v8h*)(p + 16 + 8 * half);
  return __builtin_shufflevector(lo, hi, 0, 1, 2, 3, 4, 5, 6, 7,
                                 8, 9, 10, 11, 12, 13, 14, 15);
}

// One wave computes a 32x32 f32 block of C = A(32xK) * B(Kx32).
// Abase: row-major, points at row tm.  BTbase: row-major B^T, points at row tn.
__device__ __forceinline__ void wmma_block32(const _Float16* __restrict__ Abase, int lda,
                                             const _Float16* __restrict__ BTbase, int ldbt,
                                             int K, v8f c[2][2]) {
  const int lane = threadIdx.x & 31;
  const int half = lane >> 4;
  const int m    = lane & 15;
  const _Float16* a0p = Abase  + (size_t)m * lda;
  const _Float16* a1p = a0p    + (size_t)16 * lda;
  const _Float16* b0p = BTbase + (size_t)m * ldbt;
  const _Float16* b1p = b0p    + (size_t)16 * ldbt;
  for (int k0 = 0; k0 < K; k0 += 32) {
    v16h a0 = load_frag_A (a0p + k0, half);
    v16h a1 = load_frag_A (a1p + k0, half);
    v16h b0 = load_frag_BT(b0p + k0, half);
    v16h b1 = load_frag_BT(b1p + k0, half);
    c[0][0] = __builtin_amdgcn_wmma_f32_16x16x32_f16(false, a0, false, b0, (short)0, c[0][0], false, false);
    c[0][1] = __builtin_amdgcn_wmma_f32_16x16x32_f16(false, a0, false, b1, (short)0, c[0][1], false, false);
    c[1][0] = __builtin_amdgcn_wmma_f32_16x16x32_f16(false, a1, false, b0, (short)0, c[1][0], false, false);
    c[1][1] = __builtin_amdgcn_wmma_f32_16x16x32_f16(false, a1, false, b1, (short)0, c[1][1], false, false);
  }
}

// ---------------------------------------------------------------------------
// h = atoms @ W (f32 math), stored TRANSPOSED as f16 hT[64][1024] per graph
// for the final WMMA GEMM; plus src_i = h_i.a[0:64], dst_i = h_i.a[64:128].
// One 64-thread block per node.
// ---------------------------------------------------------------------------
__global__ void k_build_h(const float* __restrict__ atoms, const float* __restrict__ W,
                          const float* __restrict__ avec, _Float16* __restrict__ hT,
                          float* __restrict__ srcv, float* __restrict__ dstv) {
  __shared__ float at[FIN];
  __shared__ float red[FOUT];
  const int node = blockIdx.x;          // b*GN + i
  const int b = node >> 10, i = node & (GN - 1);
  const int f = threadIdx.x;            // 0..63
  at[f]      = atoms[(size_t)node * FIN + f];
  at[f + 64] = atoms[(size_t)node * FIN + f + 64];
  __syncthreads();
  float acc = 0.f;
#pragma unroll 8
  for (int k = 0; k < FIN; ++k) acc = fmaf(at[k], W[k * FOUT + f], acc);
  hT[((size_t)b * FOUT + f) * GN + i] = (_Float16)acc;

  red[f] = acc * avec[f];
  __syncthreads();
  for (int s = 32; s > 0; s >>= 1) { if (f < s) red[f] += red[f + s]; __syncthreads(); }
  if (f == 0) srcv[node] = red[0];
  __syncthreads();
  red[f] = acc * avec[FOUT + f];
  __syncthreads();
  for (int s = 32; s > 0; s >>= 1) { if (f < s) red[f] += red[f + s]; __syncthreads(); }
  if (f == 0) dstv[node] = red[0];
}

// Scatter adjacency into BOTH layouts: A[i][j]=1 and AT[j][i]=1.
// Duplicate edges write the same value: benign race.
__global__ void k_scatter(const int* __restrict__ edges, _Float16* __restrict__ A,
                          _Float16* __restrict__ AT) {
  const int t = blockIdx.x * blockDim.x + threadIdx.x;
  if (t >= GB * GN * GD) return;
  const int bi = t / GD;                // b*GN + i
  const int b = bi >> 10, i = bi & (GN - 1);
  const int idx = edges[t];
  const size_t gbase = (size_t)b * GN * GN;
  A [gbase + (size_t)i * GN + idx] = (_Float16)1.0f;
  AT[gbase + (size_t)idx * GN + i] = (_Float16)1.0f;
}

// Generic 32x32-tile store helper index: row = tm + 16*ti + r + 8*half
#define TILE_STORE(body)                                                    \
  const int lane = threadIdx.x & 31, half = lane >> 4, col = lane & 15;     \
  _Pragma("unroll") for (int ti = 0; ti < 2; ++ti)                          \
  _Pragma("unroll") for (int tj = 0; tj < 2; ++tj)                          \
  _Pragma("unroll") for (int r = 0; r < 8; ++r) {                           \
    const int row = tm + 16 * ti + r + 8 * half;                            \
    const int cc  = tn + 16 * tj + col;                                     \
    body                                                                    \
  }

// P2 = A @ A   (f16 out; integer 2-path counts <= 8: exact)
__global__ void k_gemm_AA(const _Float16* __restrict__ A, const _Float16* __restrict__ AT,
                          _Float16* __restrict__ P2) {
  const int b = blockIdx.z;
  const size_t gbase = (size_t)b * GN * GN;
  const int tm = blockIdx.x << 5, tn = blockIdx.y << 5;
  v8f c[2][2] = {};
  wmma_block32(A + gbase + (size_t)tm * GN, GN, AT + gbase + (size_t)tn * GN, GN, GN, c);
  _Float16* Pb = P2 + gbase;
  TILE_STORE({ Pb[(size_t)row * GN + cc] = (_Float16)c[ti][tj][r]; })
}

// S = A + P2 + P2 @ A   (fused epilogue; integer values <= 73: exact in f16)
__global__ void k_gemm_S(const _Float16* __restrict__ A, const _Float16* __restrict__ AT,
                         const _Float16* __restrict__ P2, _Float16* __restrict__ S) {
  const int b = blockIdx.z;
  const size_t gbase = (size_t)b * GN * GN;
  const int tm = blockIdx.x << 5, tn = blockIdx.y << 5;
  v8f c[2][2] = {};
  wmma_block32(P2 + gbase + (size_t)tm * GN, GN, AT + gbase + (size_t)tn * GN, GN, GN, c);
  const _Float16* Ab = A + gbase;
  const _Float16* Pb = P2 + gbase;
  _Float16* Sb = S + gbase;
  TILE_STORE({
    const size_t idx = (size_t)row * GN + cc;
    Sb[idx] = (_Float16)(c[ti][tj][r] + (float)Ab[idx] + (float)Pb[idx]);
  })
}

// One wave per row: dense softmax over S, sparse masked softmax for attention
// (A > 0 test => exact zeros off the adjacency), blend 0.5/0.5, write f16
// in place over S.
__global__ void k_blend(_Float16* __restrict__ S, const _Float16* __restrict__ A,
                        const float* __restrict__ srcv, const float* __restrict__ dstv) {
  const int row = blockIdx.x, b = blockIdx.y;
  const int lane = threadIdx.x;
  const size_t base = ((size_t)b * GN + row) * GN;
  const float si = srcv[b * GN + row];
  const float* drow = dstv + b * GN;

  float ms = -3.0e38f, ma = -3.0e38f;
  for (int j = lane; j < GN; j += 32) {
    ms = fmaxf(ms, (float)S[base + j]);
    if ((float)A[base + j] > 0.f) {
      const float x = si + drow[j];
      ma = fmaxf(ma, x >= 0.f ? x : LALPHA * x);
    }
  }
  for (int off = 16; off > 0; off >>= 1) {
    ms = fmaxf(ms, __shfl_xor(ms, off, 32));
    ma = fmaxf(ma, __shfl_xor(ma, off, 32));
  }
  float zs = 0.f, za = 0.f;
  for (int j = lane; j < GN; j += 32) {
    zs += expf((float)S[base + j] - ms);
    if ((float)A[base + j] > 0.f) {
      const float x = si + drow[j];
      const float e = x >= 0.f ? x : LALPHA * x;
      za += expf(e - ma);
    }
  }
  for (int off = 16; off > 0; off >>= 1) {
    zs += __shfl_xor(zs, off, 32);
    za += __shfl_xor(za, off, 32);
  }
  const float rzs = 0.5f / zs, rza = 0.5f / za;   // every row has >=1 neighbor
  for (int j = lane; j < GN; j += 32) {
    float v = expf((float)S[base + j] - ms) * rzs;
    if ((float)A[base + j] > 0.f) {
      const float x = si + drow[j];
      const float e = x >= 0.f ? x : LALPHA * x;
      v += expf(e - ma) * rza;
    }
    S[base + j] = (_Float16)v;
  }
}

// out = elu(blended @ h), f32 output. B operand = hT (transposed h).
__global__ void k_gemm_out(const _Float16* __restrict__ S, const _Float16* __restrict__ hT,
                           float* __restrict__ out) {
  const int b = blockIdx.z;
  const _Float16* Sb = S  + (size_t)b * GN * GN;
  const _Float16* hb = hT + (size_t)b * FOUT * GN;
  const int tm = blockIdx.x << 5, tn = blockIdx.y << 5;
  v8f c[2][2] = {};
  wmma_block32(Sb + (size_t)tm * GN, GN, hb + (size_t)tn * GN, GN, GN, c);
  float* ob = out + (size_t)b * GN * FOUT;
  TILE_STORE({
    const float v = c[ti][tj][r];
    ob[(size_t)row * FOUT + cc] = v > 0.f ? v : (expf(v) - 1.f);
  })
}

// ---------------------------------------------------------------------------
extern "C" void kernel_launch(void* const* d_in, const int* in_sizes, int n_in,
                              void* d_out, int out_size, void* d_ws, size_t ws_size,
                              hipStream_t stream) {
  (void)in_sizes; (void)n_in; (void)out_size; (void)ws_size;
  const float* atoms = (const float*)d_in[0];   // [16,1024,128] f32
  const int*   edges = (const int*)d_in[1];     // [16,1024,8]   int
  const float* W     = (const float*)d_in[2];   // [128,64]      f32
  const float* avec  = (const float*)d_in[3];   // [128,1]       f32
  float* out = (float*)d_out;                   // [16,1024,64]  f32

  char* ws = (char*)d_ws;
  _Float16* A   = (_Float16*)ws; ws += (size_t)GB * GN * GN * sizeof(_Float16);
  _Float16* AT  = (_Float16*)ws; ws += (size_t)GB * GN * GN * sizeof(_Float16);
  _Float16* P2  = (_Float16*)ws; ws += (size_t)GB * GN * GN * sizeof(_Float16);
  _Float16* S   = (_Float16*)ws; ws += (size_t)GB * GN * GN * sizeof(_Float16);
  _Float16* hT  = (_Float16*)ws; ws += (size_t)GB * FOUT * GN * sizeof(_Float16);
  float* srcv = (float*)ws; ws += (size_t)GB * GN * sizeof(float);
  float* dstv = (float*)ws; ws += (size_t)GB * GN * sizeof(float);

  hipMemsetAsync(A,  0, (size_t)GB * GN * GN * sizeof(_Float16), stream);
  hipMemsetAsync(AT, 0, (size_t)GB * GN * GN * sizeof(_Float16), stream);

  k_build_h<<<GB * GN, FOUT, 0, stream>>>(atoms, W, avec, hT, srcv, dstv);
  k_scatter<<<(GB * GN * GD + 255) / 256, 256, 0, stream>>>(edges, A, AT);

  dim3 gSq(GN / 32, GN / 32, GB);
  k_gemm_AA<<<gSq, 32, 0, stream>>>(A, AT, P2);        // P2 = A@A
  k_gemm_S <<<gSq, 32, 0, stream>>>(A, AT, P2, S);     // S  = A + P2 + P2@A

  k_blend<<<dim3(GN, GB), 32, 0, stream>>>(S, A, srcv, dstv);  // S <- blended

  k_gemm_out<<<dim3(GN / 32, FOUT / 32, GB), 32, 0, stream>>>(S, hT, out);
}
